// TSKIPS_EncoderLayer_61400852463767
// MI455X (gfx1250) — matrix operand order repaired and checked by reading, not compiled, not run on hardware
//
#include <hip/hip_runtime.h>
#include <hip/hip_bf16.h>
#include <cstdint>

typedef __attribute__((ext_vector_type(16))) _Float16 v16h;
typedef __attribute__((ext_vector_type(8)))  _Float16 v8h;
typedef __attribute__((ext_vector_type(8)))  float    v8f;

#ifndef USE_ASYNC_LDS
#define USE_ASYNC_LDS 1     // gfx1250 GLOBAL_LOAD_ASYNC_TO_LDS path (ASYNCcnt)
#endif

// ---------------- problem constants ----------------
constexpr int    Dm  = 256;
constexpr int    FF  = 1024;
constexpr int    Bb  = 8;
constexpr int    Nn  = 1024;
constexpr int    Tt  = 16;
constexpr int    Mrows = Bb * Nn;                 // 8192 rows per GEMM
constexpr size_t R   = (size_t)Bb * Nn * Dm;      // 2,097,152
constexpr size_t RF  = (size_t)Bb * Nn * FF;      // 8,388,608
constexpr float  DECAYF = 0.5f;
constexpr float  VTHF   = 1.0f;
constexpr float  EPSF   = 1e-5f;

// ---------------- workspace layout (bytes) ----------------
// Weights are stored TRANSPOSED in f16: Wt[n*K + k] = (f16)W[k*N + n]
constexpr size_t OFF_WQ   = 0;
constexpr size_t OFF_WK   = OFF_WQ  + (size_t)Dm*Dm*2;
constexpr size_t OFF_WV   = OFF_WK  + (size_t)Dm*Dm*2;
constexpr size_t OFF_WO   = OFF_WV  + (size_t)Dm*Dm*2;
constexpr size_t OFF_WP   = OFF_WO  + (size_t)Dm*Dm*2;
constexpr size_t OFF_WE   = OFF_WP  + (size_t)Dm*Dm*2;
constexpr size_t OFF_W1   = OFF_WE  + (size_t)Dm*Dm*2;
constexpr size_t OFF_W2   = OFF_W1  + (size_t)Dm*FF*2;
constexpr size_t OFF_VQ   = OFF_W2  + (size_t)FF*Dm*2;   // membranes start (contiguous, zeroed)
constexpr size_t OFF_VK   = OFF_VQ  + R*4;
constexpr size_t OFF_VV   = OFF_VK  + R*4;
constexpr size_t OFF_V2M  = OFF_VV  + R*4;
constexpr size_t OFF_V1M  = OFF_V2M + R*4;
constexpr size_t OFF_X16  = OFF_V1M + RF*4;              // membranes end
constexpr size_t OFF_Q16  = OFF_X16  + R*2;
constexpr size_t OFF_K16  = OFF_Q16  + R*2;
constexpr size_t OFF_VS16 = OFF_K16  + R*2;
constexpr size_t OFF_A16  = OFF_VS16 + R*2;
constexpr size_t OFF_X116 = OFF_A16  + R*2;
constexpr size_t OFF_R16  = OFF_X116 + R*2;
constexpr size_t OFF_PC16 = OFF_R16  + R*2;
constexpr size_t OFF_H16  = OFF_PC16 + R*2;
constexpr size_t OFF_FS16 = OFF_H16  + RF*2;
constexpr size_t OFF_YO   = OFF_FS16 + R*2;
constexpr size_t OFF_X1   = OFF_YO   + R*4;
constexpr size_t OFF_PC   = OFF_X1   + R*4;
constexpr size_t OFF_KV   = OFF_PC   + R*4;
constexpr size_t MEMB_FLOATS = 4*R + RF;                 // floats to zero each launch

// ---------------- async copy helpers (CDNA5) ----------------
__device__ __forceinline__ void async_b128(const _Float16* __restrict__ g, _Float16* l) {
#if USE_ASYNC_LDS
  // GLOBAL_LOAD_ASYNC_TO_LDS_B128: per-lane 16B global -> LDS, tracked by ASYNCcnt.
  unsigned            lds = (unsigned)(uintptr_t)l;        // low 32 bits = LDS byte addr
  unsigned long long  ga  = (unsigned long long)(uintptr_t)g;
  asm volatile("global_load_async_to_lds_b128 %0, %1, off" :: "v"(lds), "v"(ga) : "memory");
#else
  *(v8h*)l = *(const v8h*)g;
#endif
}

__device__ __forceinline__ void wait_async_lds() {
#if USE_ASYNC_LDS
#if defined(__has_builtin)
#if __has_builtin(__builtin_amdgcn_s_wait_asynccnt)
  __builtin_amdgcn_s_wait_asynccnt(0);
#else
  asm volatile("s_wait_asynccnt 0x0" ::: "memory");
#endif
#else
  asm volatile("s_wait_asynccnt 0x0" ::: "memory");
#endif
#endif
}

// ---------------- elementwise helpers ----------------
__global__ void zero_f32(float* __restrict__ p, size_t n) {
  size_t i  = (size_t)blockIdx.x * blockDim.x + threadIdx.x;
  size_t st = (size_t)gridDim.x * blockDim.x;
  for (; i < n; i += st) p[i] = 0.0f;
}

__global__ void cvt_f32_f16(const float* __restrict__ s, _Float16* __restrict__ d, size_t n) {
  size_t i  = (size_t)blockIdx.x * blockDim.x + threadIdx.x;
  size_t st = (size_t)gridDim.x * blockDim.x;
  for (; i < n; i += st) d[i] = (_Float16)s[i];
}

// Wt[n*K + k] = (f16) W[k*N + n]  -- 32x32 LDS-tiled transpose + convert.
__global__ void cvt_transpose_f16(const float* __restrict__ w, _Float16* __restrict__ wt,
                                  int K, int N) {
  __shared__ float tile[32][33];
  const int k0 = blockIdx.y * 32;
  const int n0 = blockIdx.x * 32;
  const int tx = threadIdx.x & 31;
  const int ty = threadIdx.x >> 5;
  #pragma unroll
  for (int i = ty; i < 32; i += 8)
    tile[i][tx] = w[(size_t)(k0 + i) * N + n0 + tx];
  __syncthreads();
  #pragma unroll
  for (int i = ty; i < 32; i += 8)
    wt[(size_t)(n0 + i) * K + k0 + tx] = (_Float16)tile[tx][i];
}

// kv[b,h,d] = sum_n k[b,n,h*32+d] * v[b,n,h*32+d]; grid = B*H blocks of 256
__global__ void kv_reduce(const _Float16* __restrict__ k16, const _Float16* __restrict__ v16,
                          float* __restrict__ kv) {
  __shared__ float red[256];
  const int b = blockIdx.x >> 3;
  const int h = blockIdx.x & 7;
  const int d = threadIdx.x & 31;
  const int g = threadIdx.x >> 5;        // 8 partial groups
  float sum = 0.0f;
  for (int n = g; n < Nn; n += 8) {
    size_t idx = ((size_t)b * Nn + n) * Dm + h * 32 + d;
    sum += (float)k16[idx] * (float)v16[idx];
  }
  red[threadIdx.x] = sum;
  __syncthreads();
  if (g == 0) {
    float s = 0.0f;
    #pragma unroll
    for (int gg = 0; gg < 8; ++gg) s += red[gg * 32 + d];
    kv[(b * 8 + h) * 32 + d] = s;
  }
}

// a16[b,n,c] = q[b,n,c] * kv[b, c/32, c%32]
__global__ void scale_q(const _Float16* __restrict__ q16, const float* __restrict__ kv,
                        _Float16* __restrict__ a16, size_t n) {
  size_t i  = (size_t)blockIdx.x * blockDim.x + threadIdx.x;
  size_t st = (size_t)gridDim.x * blockDim.x;
  for (; i < n; i += st) {
    int c = (int)(i & (Dm - 1));
    int b = (int)(i >> 18);              // i / (Nn*Dm)
    a16[i] = (_Float16)((float)q16[i] * kv[(b * 8 + (c >> 5)) * 32 + (c & 31)]);
  }
}

// x1 = LN(a + b), both f32; writes f32 + f16 copies. One 256-wide row per block.
__global__ void ln_add_f32(const float* __restrict__ a, const float* __restrict__ bsum,
                           const float* __restrict__ g, const float* __restrict__ be,
                           float* __restrict__ xo, _Float16* __restrict__ xo16) {
  __shared__ float red[256];
  const int c = threadIdx.x;
  const size_t idx = (size_t)blockIdx.x * Dm + c;
  float u = a[idx] + bsum[idx];
  red[c] = u; __syncthreads();
  for (int s = 128; s > 0; s >>= 1) { if (c < s) red[c] += red[c + s]; __syncthreads(); }
  float m = red[0] * (1.0f / Dm); __syncthreads();
  float dlt = u - m;
  red[c] = dlt * dlt; __syncthreads();
  for (int s = 128; s > 0; s >>= 1) { if (c < s) red[c] += red[c + s]; __syncthreads(); }
  float var = red[0] * (1.0f / Dm);
  float xn = dlt * rsqrtf(var + EPSF) * g[c] + be[c];
  xo[idx] = xn;
  xo16[idx] = (_Float16)xn;
}

// out = LN(a + (float)b16); writes f32 output only (final per-step result)
__global__ void ln_add_f16(const float* __restrict__ a, const _Float16* __restrict__ b16,
                           const float* __restrict__ g, const float* __restrict__ be,
                           float* __restrict__ xo) {
  __shared__ float red[256];
  const int c = threadIdx.x;
  const size_t idx = (size_t)blockIdx.x * Dm + c;
  float u = a[idx] + (float)b16[idx];
  red[c] = u; __syncthreads();
  for (int s = 128; s > 0; s >>= 1) { if (c < s) red[c] += red[c + s]; __syncthreads(); }
  float m = red[0] * (1.0f / Dm); __syncthreads();
  float dlt = u - m;
  red[c] = dlt * dlt; __syncthreads();
  for (int s = 128; s > 0; s >>= 1) { if (c < s) red[c] += red[c + s]; __syncthreads(); }
  float var = red[0] * (1.0f / Dm);
  xo[idx] = dlt * rsqrtf(var + EPSF) * g[c] + be[c];
}

// ---------------- WMMA GEMM (f16 in, f32 accum), fused epilogues ----------------
// Block: 256 threads = 8 waves (4 in M x 2 in N). Block tile: 128(M) x 128(N).
// Wave tile: 32(M) x 64(N) = 8 accumulators; K staged 32 at a time, DOUBLE-BUFFERED
// with async global->LDS copies overlapping the WMMAs (ASYNCcnt synchronized).
// Weights pre-transposed (Wt[n*Kd + k]) so A and B tiles stage/load as b128s.
// MODE 0: f32io[idx] = acc + bias
// MODE 1: LIF: v = mem*DECAY + acc + bias; spike -> f16o, mem updated in place
// MODE 2: f16o[idx] = aux[idx] - (acc + bias)
// MODE 3: p = aux[idx] + acc + bias; f32io = p; f16o = (f16)p
constexpr int LS_STRIDE = 40;           // 32 + 8-half pad (20 words, conflict-free)
constexpr int AS_TILE = 128 * LS_STRIDE;
constexpr int BS_TILE = 128 * LS_STRIDE;

__device__ __forceinline__ void stage_tiles(const _Float16* __restrict__ A,
                                            const _Float16* __restrict__ Wt,
                                            _Float16* Asb, _Float16* Bsb,
                                            int m0, int n0, int k0, int Kd, int tid) {
#if USE_ASYNC_LDS
  #pragma unroll
  for (int it = 0; it < 2; ++it) {
    int lin = (tid + it * 256) * 8;
    int r = lin >> 5, c = lin & 31;     // 128 rows x 32 k-halves
    async_b128(A + (size_t)(m0 + r) * Kd + k0 + c, Asb + r * LS_STRIDE + c);
  }
  #pragma unroll
  for (int it = 0; it < 2; ++it) {
    int lin = (tid + it * 256) * 8;
    int r = lin >> 5, c = lin & 31;     // 128 cols x 32 k-halves
    async_b128(Wt + (size_t)(n0 + r) * Kd + k0 + c, Bsb + r * LS_STRIDE + c);
  }
#else
  v8h abuf[2], bbuf[2];
  #pragma unroll
  for (int it = 0; it < 2; ++it) {
    int lin = (tid + it * 256) * 8;
    abuf[it] = *(const v8h*)(A + (size_t)(m0 + (lin >> 5)) * Kd + k0 + (lin & 31));
  }
  #pragma unroll
  for (int it = 0; it < 2; ++it) {
    int lin = (tid + it * 256) * 8;
    bbuf[it] = *(const v8h*)(Wt + (size_t)(n0 + (lin >> 5)) * Kd + k0 + (lin & 31));
  }
  #pragma unroll
  for (int it = 0; it < 2; ++it) {
    int lin = (tid + it * 256) * 8;
    *(v8h*)(Asb + (lin >> 5) * LS_STRIDE + (lin & 31)) = abuf[it];
  }
  #pragma unroll
  for (int it = 0; it < 2; ++it) {
    int lin = (tid + it * 256) * 8;
    *(v8h*)(Bsb + (lin >> 5) * LS_STRIDE + (lin & 31)) = bbuf[it];
  }
#endif
}

template <int MODE>
__global__ void __launch_bounds__(256)
gemm_wmma(const _Float16* __restrict__ A, const _Float16* __restrict__ Wt,
          const float* __restrict__ bias, const float* __restrict__ aux,
          float* __restrict__ f32io, _Float16* __restrict__ f16o,
          int Nd, int Kd) {
  __shared__ _Float16 As[2 * AS_TILE];   // As[m][k], double buffered
  __shared__ _Float16 Bs[2 * BS_TILE];   // Bs[n][k]  (K-major), double buffered

  const int tid  = threadIdx.x;
  const int lane = tid & 31;
  const int wave = tid >> 5;
  const int wm   = wave >> 1;            // 0..3  (M position)
  const int wn   = wave & 1;             // 0..1  (N position)
  const int m0   = blockIdx.y * 128;
  const int n0   = blockIdx.x * 128;

  v8f acc[2][4] = {};

  // prologue: stage tile 0 into buffer 0
  stage_tiles(A, Wt, As, Bs, m0, n0, 0, Kd, tid);
  wait_async_lds();
  __syncthreads();

  int p = 0;
  for (int k0 = 0; k0 < Kd; k0 += 32) {
    const _Float16* Asb = As + p * AS_TILE;
    const _Float16* Bsb = Bs + p * BS_TILE;

    if (k0 + 32 < Kd)   // kick off next stage while we compute on this one
      stage_tiles(A, Wt, As + (1 - p) * AS_TILE, Bs + (1 - p) * BS_TILE,
                  m0, n0, k0 + 32, Kd, tid);

    // A fragments (two 16-row tiles) per ISA 16-bit A 16x32 layout:
    // lanes 0-15 : row=lane,    K = {0..7, 16..23}
    // lanes 16-31: row=lane-16, K = {8..15, 24..31}
    v16h af[2];
    #pragma unroll
    for (int s = 0; s < 2; ++s) {
      const int ar = wm * 32 + s * 16 + (lane & 15);
      const int ak = (lane >> 4) << 3;
      v8h alo = *(const v8h*)(Asb + ar * LS_STRIDE + ak);
      v8h ahi = *(const v8h*)(Asb + ar * LS_STRIDE + ak + 16);
      af[s] = __builtin_shufflevector(alo, ahi,
          0, 1, 2, 3, 4, 5, 6, 7, 8, 9, 10, 11, 12, 13, 14, 15);
    }
    // B fragments (32x16): lanes 0-15 col=n K={0..7,16..23}; lanes 16-31 K={8..15,24..31}.
    v16h bf[4];
    #pragma unroll
    for (int j = 0; j < 4; ++j) {
      const int bn = wn * 64 + j * 16 + (lane & 15);
      const int bk = (lane >> 4) << 3;
      v8h blo = *(const v8h*)(Bsb + bn * LS_STRIDE + bk);
      v8h bhi = *(const v8h*)(Bsb + bn * LS_STRIDE + bk + 16);
      bf[j] = __builtin_shufflevector(blo, bhi,
          0, 1, 2, 3, 4, 5, 6, 7, 8, 9, 10, 11, 12, 13, 14, 15);
    }
    // 8 back-to-back WMMAs per stage
    #pragma unroll
    for (int s = 0; s < 2; ++s)
      #pragma unroll
      for (int j = 0; j < 4; ++j)
        acc[s][j] = __builtin_amdgcn_wmma_f32_16x16x32_f16(
            false, af[s], false, bf[j], (short)0, acc[s][j], false, false);

    wait_async_lds();     // next-stage copies done (overlapped with WMMAs above)
    __syncthreads();
    p ^= 1;
  }

  // Epilogue: C/D layout — VGPR i: lanes0-15 (M=i, N=lane), lanes16-31 (M=8+i, N=lane-16)
  #pragma unroll
  for (int s = 0; s < 2; ++s) {
    const int rbase = m0 + wm * 32 + s * 16 + ((lane >> 4) << 3);
    #pragma unroll
    for (int j = 0; j < 4; ++j) {
      const int c = n0 + wn * 64 + j * 16 + (lane & 15);
      const float bb = bias[c];
      #pragma unroll
      for (int i = 0; i < 8; ++i) {
        const size_t idx = (size_t)(rbase + i) * Nd + c;
        const float y = acc[s][j][i] + bb;
        if (MODE == 0) {
          f32io[idx] = y;
        } else if (MODE == 1) {
          float v = f32io[idx] * DECAYF + y;
          float sp = (v >= VTHF) ? 1.0f : 0.0f;
          f32io[idx] = v - sp * VTHF;
          f16o[idx]  = (_Float16)sp;
        } else if (MODE == 2) {
          f16o[idx] = (_Float16)(aux[idx] - y);
        } else {
          float ppc = aux[idx] + y;
          f32io[idx] = ppc;
          f16o[idx]  = (_Float16)ppc;
        }
      }
    }
  }
}

// ---------------- launcher ----------------
extern "C" void kernel_launch(void* const* d_in, const int* in_sizes, int n_in,
                              void* d_out, int out_size, void* d_ws, size_t ws_size,
                              hipStream_t stream) {
  (void)in_sizes; (void)n_in; (void)out_size; (void)ws_size;

  const float* src = (const float*)d_in[0];
  const float* Wq  = (const float*)d_in[1];   const float* bq  = (const float*)d_in[2];
  const float* Wk  = (const float*)d_in[3];   const float* bk  = (const float*)d_in[4];
  const float* Wv  = (const float*)d_in[5];   const float* bv  = (const float*)d_in[6];
  const float* Wo  = (const float*)d_in[7];   const float* bo  = (const float*)d_in[8];
  const float* Wp  = (const float*)d_in[9];   const float* bp  = (const float*)d_in[10];
  const float* We  = (const float*)d_in[11];  const float* be  = (const float*)d_in[12];
  const float* W1  = (const float*)d_in[13];  const float* b1  = (const float*)d_in[14];
  const float* W2  = (const float*)d_in[15];  const float* b2  = (const float*)d_in[16];
  const float* g1  = (const float*)d_in[17];  const float* bl1 = (const float*)d_in[18];
  const float* g2  = (const float*)d_in[19];  const float* bl2 = (const float*)d_in[20];
  float* out = (float*)d_out;

  char* ws = (char*)d_ws;
  _Float16* wq16 = (_Float16*)(ws + OFF_WQ);
  _Float16* wk16 = (_Float16*)(ws + OFF_WK);
  _Float16* wv16 = (_Float16*)(ws + OFF_WV);
  _Float16* wo16 = (_Float16*)(ws + OFF_WO);
  _Float16* wp16 = (_Float16*)(ws + OFF_WP);
  _Float16* we16 = (_Float16*)(ws + OFF_WE);
  _Float16* w116 = (_Float16*)(ws + OFF_W1);
  _Float16* w216 = (_Float16*)(ws + OFF_W2);
  float*    vq   = (float*)(ws + OFF_VQ);
  float*    vk   = (float*)(ws + OFF_VK);
  float*    vv   = (float*)(ws + OFF_VV);
  float*    v2m  = (float*)(ws + OFF_V2M);
  float*    v1m  = (float*)(ws + OFF_V1M);
  _Float16* x16  = (_Float16*)(ws + OFF_X16);
  _Float16* q16  = (_Float16*)(ws + OFF_Q16);
  _Float16* k16  = (_Float16*)(ws + OFF_K16);
  _Float16* vs16 = (_Float16*)(ws + OFF_VS16);
  _Float16* a16  = (_Float16*)(ws + OFF_A16);
  _Float16* x116 = (_Float16*)(ws + OFF_X116);
  _Float16* r16  = (_Float16*)(ws + OFF_R16);
  _Float16* pc16 = (_Float16*)(ws + OFF_PC16);
  _Float16* h16  = (_Float16*)(ws + OFF_H16);
  _Float16* fs16 = (_Float16*)(ws + OFF_FS16);
  float*    yo   = (float*)(ws + OFF_YO);
  float*    x1   = (float*)(ws + OFF_X1);
  float*    pc   = (float*)(ws + OFF_PC);
  float*    kv   = (float*)(ws + OFF_KV);

  // one-time-per-call prep: transposed f16 weights + zeroed membranes (deterministic)
  const dim3 blk(256);
  const dim3 tD(Dm / 32, Dm / 32);      // (8, 8)
  const dim3 t1(FF / 32, Dm / 32);      // N=FF, K=Dm -> (32, 8)
  const dim3 t2(Dm / 32, FF / 32);      // N=Dm, K=FF -> (8, 32)
  cvt_transpose_f16<<<tD, blk, 0, stream>>>(Wq, wq16, Dm, Dm);
  cvt_transpose_f16<<<tD, blk, 0, stream>>>(Wk, wk16, Dm, Dm);
  cvt_transpose_f16<<<tD, blk, 0, stream>>>(Wv, wv16, Dm, Dm);
  cvt_transpose_f16<<<tD, blk, 0, stream>>>(Wo, wo16, Dm, Dm);
  cvt_transpose_f16<<<tD, blk, 0, stream>>>(Wp, wp16, Dm, Dm);
  cvt_transpose_f16<<<tD, blk, 0, stream>>>(We, we16, Dm, Dm);
  cvt_transpose_f16<<<t1, blk, 0, stream>>>(W1, w116, Dm, FF);
  cvt_transpose_f16<<<t2, blk, 0, stream>>>(W2, w216, FF, Dm);
  zero_f32<<<2048, 256, 0, stream>>>(vq, MEMB_FLOATS);

  const dim3 gD(Dm / 128, Mrows / 128);   // (2, 64)
  const dim3 gF(FF / 128, Mrows / 128);   // (8, 64)

  for (int t = 0; t < Tt; ++t) {
    const float* xt = src + (size_t)t * R;

    cvt_f32_f16<<<2048, blk, 0, stream>>>(xt, x16, R);

    // spiking Q/K/V projections (GEMM + LIF, membranes in place)
    gemm_wmma<1><<<gD, blk, 0, stream>>>(x16, wq16, bq, nullptr, vq, q16, Dm, Dm);
    gemm_wmma<1><<<gD, blk, 0, stream>>>(x16, wk16, bk, nullptr, vk, k16, Dm, Dm);
    gemm_wmma<1><<<gD, blk, 0, stream>>>(x16, wv16, bv, nullptr, vv, vs16, Dm, Dm);

    // linear attention: kv = sum_n k*v ; a = q * kv ; attn = a @ Wo + bo
    kv_reduce<<<Bb * 8, blk, 0, stream>>>(k16, vs16, kv);
    scale_q<<<2048, blk, 0, stream>>>(q16, kv, a16, R);
    gemm_wmma<0><<<gD, blk, 0, stream>>>(a16, wo16, bo, nullptr, yo, nullptr, Dm, Dm);

    // x1 = LN(x + attn)
    ln_add_f32<<<Mrows, blk, 0, stream>>>(xt, yo, g1, bl1, x1, x116);

    // predictive coding: r = x1 - (x1@Wp+bp); pc = x1 + r@We + be
    gemm_wmma<2><<<gD, blk, 0, stream>>>(x116, wp16, bp, x1, nullptr, r16, Dm, Dm);
    gemm_wmma<3><<<gD, blk, 0, stream>>>(r16, we16, be, x1, pc, pc16, Dm, Dm);

    // spiking FFN
    gemm_wmma<1><<<gF, blk, 0, stream>>>(pc16, w116, b1, nullptr, v1m, h16, FF, Dm);
    gemm_wmma<1><<<gD, blk, 0, stream>>>(h16, w216, b2, nullptr, v2m, fs16, Dm, FF);

    // out[t] = LN(pc + f)
    ln_add_f16<<<Mrows, blk, 0, stream>>>(pc, fs16, g2, bl2, out + (size_t)t * R);
  }
}
